// BlockAutoregressiveLinear_38431367364723
// MI455X (gfx1250) — matrix operand order repaired and checked by reading, not compiled, not run on hardware
//
#include <hip/hip_runtime.h>

// ---------------------------------------------------------------------------
// BlockAutoregressiveLinear for MI455X (gfx1250, wave32, WMMA)
//   y   = x @ Wn^T + bias        (16384 x 1024 x 1024 GEMM, split-f16 WMMA)
//   jac = log(diag 8x8 blocks)   (analytic: ls + W - log(norm))
//
// GEMM roofline: 34.4 GFLOP dense, ~128 MB HBM traffic (x read + y write;
// weights L2-resident). f32 WMMA only exists as 16x16x4 -> compute-bound.
// Split-f16 (Markidis) scheme: x = xh+xl, W = wh+wl, acc += xh*wh+xh*wl+xl*wh
// on v_wmma_f32_16x16x32_f16 keeps ~2^-20 relative error at 3x the WMMA
// issue rate of a single pass -- still ~3-4x faster than the f32 K=4 path.
// Block-triangular structure cuts K to n0b+64 (~0.53x dense work/traffic).
// ---------------------------------------------------------------------------

typedef __attribute__((ext_vector_type(16))) _Float16 v16h;
typedef __attribute__((ext_vector_type(8)))  _Float16 v8h;
typedef __attribute__((ext_vector_type(8)))  float    v8f;

#define BATCH 16384
#define DIM   1024
#define BM    128
#define BN    64
#define BK    32
#define PITCH 40          // halves per LDS row (32 + 8 pad, keeps 16B align)

static __device__ inline v16h combine8(v8h lo, v8h hi) {
  v16h r;
#pragma unroll
  for (int i = 0; i < 8; ++i) { r[i] = lo[i]; r[i + 8] = hi[i]; }
  return r;
}

// ---------------------------------------------------------------------------
// Prep: Wm = exp(W)*diag + W*tril ; norm ; Wn = exp(ls)*Wm/norm
//       split Wn -> f16 hi/lo planes; jac (diag blocks) = ls + W - log(norm)
// one block per output row
// ---------------------------------------------------------------------------
__global__ __launch_bounds__(256) void bal_prep(
    const float* __restrict__ W, const float* __restrict__ Wls,
    const int* __restrict__ dmask, const int* __restrict__ tmask,
    _Float16* __restrict__ Wh, _Float16* __restrict__ Wl,
    float* __restrict__ jac)
{
  __shared__ float red[256];
  const int r   = blockIdx.x;
  const int tid = threadIdx.x;

  float wm[4], wraw[4];
  int   dm[4];
  float p = 0.0f;
#pragma unroll
  for (int j = 0; j < 4; ++j) {
    const int c  = tid + j * 256;
    const float wv = W[r * DIM + c];
    const int d = dmask[r * DIM + c];
    const int t = tmask[r * DIM + c];
    const float m = expf(wv) * (float)d + wv * (float)t;
    wm[j] = m; wraw[j] = wv; dm[j] = d;
    p += m * m;
  }
  red[tid] = p;
  __syncthreads();
#pragma unroll
  for (int s = 128; s > 0; s >>= 1) {
    if (tid < s) red[tid] += red[tid + s];
    __syncthreads();
  }
  const float norm  = sqrtf(red[0]);
  const float ls    = Wls[r];
  const float scale = expf(ls) / norm;
  const float lnorm = logf(norm);

#pragma unroll
  for (int j = 0; j < 4; ++j) {
    const int c  = tid + j * 256;
    const float wn = wm[j] * scale;
    const _Float16 hi = (_Float16)wn;
    Wh[r * DIM + c] = hi;
    Wl[r * DIM + c] = (_Float16)(wn - (float)hi);
    if (dm[j]) {  // diagonal-block entry -> jac output
      const int blk = r >> 3;
      jac[blk * 64 + (r & 7) * 8 + (c & 7)] = ls + wraw[j] - lnorm;
    }
  }
}

// ---------------------------------------------------------------------------
// GEMM: y[m,n] = sum_k x[m,k] * Wn[n,k] + bias[n]
// Block tile 128M x 64N, 8 waves (2M x 4N), wave = 4 x (16x16) D tiles.
// Software pipeline: x tile for step k+1 is loaded into registers while the
// WMMAs for step k run, so global latency never sits inside the barrier
// critical path.
// ---------------------------------------------------------------------------
__global__ __launch_bounds__(256) void bal_gemm(
    const float* __restrict__ x,
    const _Float16* __restrict__ Wh, const _Float16* __restrict__ Wl,
    const float* __restrict__ bias,
    float* __restrict__ y)
{
  __shared__ __align__(16) _Float16 sxh[BM * PITCH];
  __shared__ __align__(16) _Float16 sxl[BM * PITCH];

  const int m0b  = (int)blockIdx.x * BM;
  const int n0b  = (int)blockIdx.y * BN;
  const int tid  = (int)threadIdx.x;
  const int lane = tid & 31;
  const int wid  = tid >> 5;
  const int wm_  = wid & 1;          // M half (0/1) -> 64 rows each
  const int wn_  = wid >> 1;         // N quarter (0..3) -> 16 cols each
  const int n0w  = n0b + wn_ * 16;
  const int m0w  = m0b + wm_ * 64;
  const int l15  = lane & 15;
  const int hiHalf = (lane >= 16);   // WMMA lane group

  const int kmax = min(DIM, n0b + BN);   // block-triangular skip

  v8f acc[4] = {};

  // x tile loader coords: thread -> (row, 16-float half-row)
  const int xr   = tid >> 1;         // 0..127
  const int xh16 = (tid & 1) * 16;   // 0 or 16
  const float* xrow = x + (size_t)(m0b + xr) * DIM + xh16;

  // ---- pipeline prologue: load x tile for k0 = 0 into registers ----
  float4 fx[4];
#pragma unroll
  for (int i = 0; i < 4; ++i) fx[i] = ((const float4*)(xrow))[i];

  for (int k0 = 0; k0 < kmax; k0 += BK) {
    // ---- split current register tile into f16 hi/lo (register-only) ----
    __align__(16) _Float16 hh[16], ll[16];
    {
      const float* fp = (const float*)&fx[0];
#pragma unroll
      for (int i = 0; i < 16; ++i) {
        const _Float16 h = (_Float16)fp[i];
        hh[i] = h;
        ll[i] = (_Float16)(fp[i] - (float)h);
      }
    }
    __syncthreads();                 // previous iteration consumers done
    {
      _Float16* dh = &sxh[xr * PITCH + xh16];
      _Float16* dl = &sxl[xr * PITCH + xh16];
      ((float4*)dh)[0] = ((const float4*)hh)[0];
      ((float4*)dh)[1] = ((const float4*)hh)[1];
      ((float4*)dl)[0] = ((const float4*)ll)[0];
      ((float4*)dl)[1] = ((const float4*)ll)[1];
    }
    __syncthreads();                 // tile visible to all waves

    // ---- preload next x tile: overlaps the WMMAs below ----
    if (k0 + BK < kmax) {
      const float* src = xrow + k0 + BK;
#pragma unroll
      for (int i = 0; i < 4; ++i) fx[i] = ((const float4*)src)[i];
    }

    // ---- B fragments (Wn^T tile): lane<16 -> col n0w+lane, K{0..7,16..23}
    //                               lane>=16 -> col n0w+lane-16, K{8..15,24..31}
    const size_t woff = (size_t)(n0w + l15) * DIM + k0 + (hiHalf ? 8 : 0);
    const v16h bh = combine8(*(const v8h*)(Wh + woff), *(const v8h*)(Wh + woff + 16));
    const v16h bl = combine8(*(const v8h*)(Wl + woff), *(const v8h*)(Wl + woff + 16));

    // ---- A fragments + 3 WMMA passes per 16-row subtile ----
#pragma unroll
    for (int s = 0; s < 4; ++s) {
      const int rowL = wm_ * 64 + s * 16 + l15;
      const int aoff = rowL * PITCH + (hiHalf ? 8 : 0);
      // lane<16: K{0..7,16..23} ; lane>=16: K{8..15,24..31}
      const v16h ah = combine8(*(const v8h*)(sxh + aoff),
                               *(const v8h*)(sxh + aoff + 16));
      const v16h al = combine8(*(const v8h*)(sxl + aoff),
                               *(const v8h*)(sxl + aoff + 16));
      acc[s] = __builtin_amdgcn_wmma_f32_16x16x32_f16(
                   false, ah, false, bh, (short)0, acc[s], false, false);
      acc[s] = __builtin_amdgcn_wmma_f32_16x16x32_f16(
                   false, ah, false, bl, (short)0, acc[s], false, false);
      acc[s] = __builtin_amdgcn_wmma_f32_16x16x32_f16(
                   false, al, false, bh, (short)0, acc[s], false, false);
    }
  }

  // ---- epilogue: bias + store (D layout: lane owns column n0w+l15,
  //      VGPR r -> row r (lanes 0-15) / row 8+r (lanes 16-31)) ----
  const int   col = n0w + l15;
  const float bv  = bias[col];
#pragma unroll
  for (int s = 0; s < 4; ++s) {
    const int rowbase = m0w + s * 16 + (hiHalf ? 8 : 0);
#pragma unroll
    for (int r = 0; r < 8; ++r)
      y[(size_t)(rowbase + r) * DIM + col] = acc[s][r] + bv;
  }
}

extern "C" void kernel_launch(void* const* d_in, const int* in_sizes, int n_in,
                              void* d_out, int out_size, void* d_ws, size_t ws_size,
                              hipStream_t stream) {
  const float* x     = (const float*)d_in[0];
  const float* W     = (const float*)d_in[1];
  const float* bias  = (const float*)d_in[2];
  const float* Wls   = (const float*)d_in[3];
  const int*   dmask = (const int*)d_in[4];
  const int*   tmask = (const int*)d_in[5];

  float* y   = (float*)d_out;
  float* jac = y + (size_t)BATCH * DIM;          // outputs concatenated flat

  _Float16* Wh = (_Float16*)d_ws;                // 2 MB
  _Float16* Wl = Wh + (size_t)DIM * DIM;         // 2 MB

  bal_prep<<<DIM, 256, 0, stream>>>(W, Wls, dmask, tmask, Wh, Wl, jac);
  bal_gemm<<<dim3(BATCH / BM, DIM / BN), 256, 0, stream>>>(x, Wh, Wl, bias, y);
}
